// RNN_23931557773385
// MI455X (gfx1250) — matrix operand order repaired and checked by reading, not compiled, not run on hardware
//
#include <hip/hip_runtime.h>

typedef _Float16 f16;
typedef __attribute__((ext_vector_type(16))) _Float16 v16h;
typedef __attribute__((ext_vector_type(8)))  _Float16 v8h;
typedef __attribute__((ext_vector_type(8)))  float    v8f;

#define B_ 32
#define T_ 200
#define E_ 32
#define H_ 128
#define V_ 8000

// ---------------------------------------------------------------------------
// 1) Vw fp32 -> fp16, kept row-major [V][H] == [N][K]  (WMMA B-fragment friendly)
// ---------------------------------------------------------------------------
__global__ void vw_to_f16(const float* __restrict__ Vw, f16* __restrict__ VwF16) {
    int i = blockIdx.x * 256 + threadIdx.x;
    if (i < V_ * H_) VwF16[i] = (f16)Vw[i];
}

// ---------------------------------------------------------------------------
// 2) ux[b,t,h] = Uw[h,:] . x[b,t,:] + Ub[h] + Wb[h]   (fp32, fully parallel)
// ---------------------------------------------------------------------------
__global__ void ux_proj(const float* __restrict__ x, const float* __restrict__ Uw,
                        const float* __restrict__ Ub, const float* __restrict__ Wb,
                        float* __restrict__ ux) {
    int i  = blockIdx.x * 256 + threadIdx.x;   // over B*T*H (exact multiple of 256)
    int h  = i & (H_ - 1);
    int bt = i >> 7;
    const float* xr = x  + (size_t)bt * E_;
    const float* ur = Uw + (size_t)h  * E_;
    float acc = Ub[h] + Wb[h];
#pragma unroll
    for (int e = 0; e < E_; ++e) acc += ur[e] * xr[e];
    ux[i] = acc;
}

// ---------------------------------------------------------------------------
// 3) Sequential Elman scan in fp32 (error compounds over T=200 -> keep fp32).
//    One thread per (batch-lane, h). WwT in dynamic LDS, conflict-free reads.
//    Emits hs as f16 [B*T][H] for the WMMA output GEMM, and the final hidden
//    state (fp32) into the tail of d_out.
// ---------------------------------------------------------------------------
__global__ void rnn_scan(const float* __restrict__ Ww, const float* __restrict__ ux,
                         f16* __restrict__ hsF16, float* __restrict__ hiddenOut) {
    extern __shared__ float smem[];
    float* sWwT = smem;              // [128][128]: sWwT[k*128+j] = Ww[j][k]
    float* hsh  = smem + H_ * H_;    // [4][128] current hidden per local batch

    int tid = threadIdx.x;           // 512 threads
    int bl  = tid >> 7;              // local batch 0..3
    int h   = tid & 127;
    int b   = blockIdx.x * 4 + bl;

    for (int idx = tid; idx < H_ * H_; idx += 512) {
        int j = idx >> 7, k = idx & 127;
        sWwT[k * H_ + j] = Ww[idx];                  // coalesced read, transposed store
    }
    hsh[bl * H_ + h] = 0.0f;
    __syncthreads();

    const float* wcol = sWwT + h;                    // stride-H_ column == Ww row h
    float hval = 0.0f;
    for (int t = 0; t < T_; ++t) {
        float acc = ux[((size_t)b * T_ + t) * H_ + h];
        const float* hrow = hsh + bl * H_;
#pragma unroll 8
        for (int k = 0; k < H_; ++k) acc += wcol[k * H_] * hrow[k]; // lanes: consec addr / broadcast
        hval = tanhf(acc);
        __syncthreads();                              // all reads of hsh done
        hsh[bl * H_ + h] = hval;
        hsF16[((size_t)b * T_ + t) * H_ + h] = (f16)hval;
        __syncthreads();                              // writes visible for next step
    }
    hiddenOut[(size_t)b * H_ + h] = hval;
}

// ---------------------------------------------------------------------------
// 4) Output projection: [6400 x 128] f16  x  [128 x 8000] f16  -> fp32 +Vb.
//    Store-BW bound (205 MB @ 23.3 TB/s ~= 8.8us floor). One wave per
//    16(M) x 64(N) strip. Per kf: 4 B fragments in DISTINCT registers so all
//    8 global_load_b128 issue in one clause, then 4 back-to-back WMMAs.
//    Output stored non-temporally (written once, never re-read) so the 205 MB
//    stream doesn't evict the L2-resident VwF16/hsF16 operands.
// ---------------------------------------------------------------------------
__global__ void __launch_bounds__(256)
out_gemm(const f16* __restrict__ A,      // hsF16 [6400][128]  (M x K)
         const f16* __restrict__ Bm,     // VwF16 [8000][128]  (N x K)
         const float* __restrict__ Vb,
         float* __restrict__ out) {      // [6400][8000]
    int wid  = blockIdx.x * 8 + (threadIdx.x >> 5);  // 0..49999
    int lane = threadIdx.x & 31;
    int half = lane >> 4;                            // lane-half selects K sub-chunk
    int l16  = lane & 15;
    int mTile  = wid / 125;                          // 0..399
    int nGroup = wid % 125;                          // 0..124
    int m0 = mTile * 16;
    int n0 = nGroup * 64;

    // A fragments: contiguous K order -> lane half picks K {+0..15} / {+16..31}
    const f16* Arow = A + (size_t)(m0 + l16) * H_;
    v16h a[4];
#pragma unroll
    for (int kf = 0; kf < 4; ++kf) {
        a[kf] = *(const v16h*)(Arow + kf * 32 + half * 16);
    }

    // B row pointers: lane = column N of each of the 4 N-subtiles
    const f16* Brow0 = Bm + (size_t)(n0 +  0 + l16) * H_ + half * 16;
    const f16* Brow1 = Bm + (size_t)(n0 + 16 + l16) * H_ + half * 16;
    const f16* Brow2 = Bm + (size_t)(n0 + 32 + l16) * H_ + half * 16;
    const f16* Brow3 = Bm + (size_t)(n0 + 48 + l16) * H_ + half * 16;

    v8f c0 = {0.f,0.f,0.f,0.f,0.f,0.f,0.f,0.f};
    v8f c1 = c0, c2 = c0, c3 = c0;

#pragma unroll
    for (int kf = 0; kf < 4; ++kf) {
        // 4 distinct B fragments -> one load clause, then 4 WMMAs overlap latency
        v16h b0 = *(const v16h*)(Brow0 + kf * 32);
        v16h b1 = *(const v16h*)(Brow1 + kf * 32);
        v16h b2 = *(const v16h*)(Brow2 + kf * 32);
        v16h b3 = *(const v16h*)(Brow3 + kf * 32);
        c0 = __builtin_amdgcn_wmma_f32_16x16x32_f16(false, a[kf], false, b0, (short)0, c0, false, false);
        c1 = __builtin_amdgcn_wmma_f32_16x16x32_f16(false, a[kf], false, b1, (short)0, c1, false, false);
        c2 = __builtin_amdgcn_wmma_f32_16x16x32_f16(false, a[kf], false, b2, (short)0, c2, false, false);
        c3 = __builtin_amdgcn_wmma_f32_16x16x32_f16(false, a[kf], false, b3, (short)0, c3, false, false);
    }

    // C/D layout: lane -> N, vgpr r -> M = m0 + 8*half + r. Non-temporal stores.
    float* obase = out + (size_t)(m0 + half * 8) * V_;
    {
        int n = n0 + l16;          float vb = Vb[n]; float* o = obase + n;
#pragma unroll
        for (int r = 0; r < 8; ++r) __builtin_nontemporal_store(c0[r] + vb, o + (size_t)r * V_);
    }
    {
        int n = n0 + 16 + l16;     float vb = Vb[n]; float* o = obase + n;
#pragma unroll
        for (int r = 0; r < 8; ++r) __builtin_nontemporal_store(c1[r] + vb, o + (size_t)r * V_);
    }
    {
        int n = n0 + 32 + l16;     float vb = Vb[n]; float* o = obase + n;
#pragma unroll
        for (int r = 0; r < 8; ++r) __builtin_nontemporal_store(c2[r] + vb, o + (size_t)r * V_);
    }
    {
        int n = n0 + 48 + l16;     float vb = Vb[n]; float* o = obase + n;
#pragma unroll
        for (int r = 0; r < 8; ++r) __builtin_nontemporal_store(c3[r] + vb, o + (size_t)r * V_);
    }
}

// ---------------------------------------------------------------------------
extern "C" void kernel_launch(void* const* d_in, const int* in_sizes, int n_in,
                              void* d_out, int out_size, void* d_ws, size_t ws_size,
                              hipStream_t stream) {
    const float* x  = (const float*)d_in[0];
    const float* Ww = (const float*)d_in[1];
    const float* Wb = (const float*)d_in[2];
    const float* Uw = (const float*)d_in[3];
    const float* Ub = (const float*)d_in[4];
    const float* Vw = (const float*)d_in[5];
    const float* Vb = (const float*)d_in[6];
    float* out = (float*)d_out;

    // Workspace layout (16B aligned): VwF16 | hsF16 | ux   (~7 MB total)
    char* ws   = (char*)d_ws;
    f16*  VwF16 = (f16*)ws;                               // V*H*2  = 2,048,000 B
    f16*  hsF16 = (f16*)(ws + 2048000);                   // B*T*H*2= 1,638,400 B
    float* ux   = (float*)(ws + 2048000 + 1638400);       // B*T*H*4= 3,276,800 B

    vw_to_f16<<<(V_ * H_ + 255) / 256, 256, 0, stream>>>(Vw, VwF16);
    ux_proj<<<(B_ * T_ * H_) / 256, 256, 0, stream>>>(x, Uw, Ub, Wb, ux);
    rnn_scan<<<B_ / 4, 512, (H_ * H_ + 4 * H_) * sizeof(float), stream>>>(
        Ww, ux, hsF16, out + (size_t)B_ * T_ * V_);
    out_gemm<<<(400 * 125) / 8, 256, 0, stream>>>(hsF16, VwF16, Vb, out);
}
